// Transformer_Decoder_77008763617516
// MI455X (gfx1250) — compile-verified, matrix operand order checked by reading, not run on hardware
//
#include <hip/hip_runtime.h>
#include <hip/hip_bf16.h>
#include <math.h>

typedef __attribute__((ext_vector_type(16))) _Float16 v16h;
typedef __attribute__((ext_vector_type(8)))  float    v8f;

// ---- model dims -------------------------------------------------------------
#define Bn    8
#define Cc    64
#define Ss    64
#define NSEG  256
#define DIM   256
#define DEPTH 4
#define HEADS 8
#define DH    64
#define MLPD  1024
#define NCLS  4
#define D2    512
#define INNER 512
#define NTOK  (Bn * NSEG)          // 2048 tokens
#define QS    (3 * INNER)          // qkv row stride = 1536

// =============================================================================
// Fragment helpers — CDNA5 16-bit A-matrix layout (ISA 7.12.2):
//   lane = hi*16 + lo;  element i of v16h:  K = (i>>3)*16 + hi*8 + (i&7), M = lo
// =============================================================================
__device__ __forceinline__ v16h frag_row_h(const _Float16* p) {
    // p already points at A[row=lo][k0 + hi*8]; two contiguous 16B chunks
    v16h r;
#pragma unroll
    for (int i = 0; i < 8; ++i) {
        r[i]     = p[i];
        r[i + 8] = p[i + 16];
    }
    return r;
}

__device__ __forceinline__ v8f wmma_f16(v16h a, v16h b, v8f c) {
    return __builtin_amdgcn_wmma_f32_16x16x32_f16(false, a, false, b, (short)0, c,
                                                  false, false);
}

// =============================================================================
// 0) init x_seg with broadcast patch bias
// =============================================================================
__global__ void init_xseg_kernel(float* __restrict__ xseg,
                                 const float* __restrict__ b_patch) {
    int idx = blockIdx.x * 256 + threadIdx.x;      // Bn*NSEG*DIM = 524288
    xseg[idx] = b_patch[idx & (DIM - 1)];
}

// =============================================================================
// 1) patch embed + superpixel segment-sum.  One block per pixel (j,h).
//    Streams the 268MB W_patch exactly once (block-wide coalesced loads).
// =============================================================================
__global__ void patch_embed_kernel(const float* __restrict__ feat2,  // [B,C,S,S]
                                   const int*   __restrict__ sp,     // [B,S,S]
                                   const float* __restrict__ W,      // [C*S*S, DIM]
                                   float* xseg) {                    // [B,NSEG,DIM]
    __shared__ float fv[Bn * Cc];
    __shared__ int   segs[Bn];
    const int pix = blockIdx.x;            // 0..4095
    const int j   = pix >> 6;
    const int hh  = pix & 63;
    const int tid = threadIdx.x;           // 256 threads

#pragma unroll
    for (int t = 0; t < 2; ++t) {
        int idx = tid + t * 256;           // b*64 + c
        int b = idx >> 6, c = idx & 63;
        fv[idx] = feat2[(((size_t)b * Cc + c) * Ss + j) * Ss + hh];
    }
    if (tid < Bn) segs[tid] = sp[((size_t)tid * Ss + j) * Ss + hh];
    __syncthreads();

    const int d = tid;
    float acc[Bn] = {0.f, 0.f, 0.f, 0.f, 0.f, 0.f, 0.f, 0.f};
    const float* wp = W + ((size_t)j * Ss + hh) * DIM + d;
    const size_t cstride = (size_t)Ss * Ss * DIM;  // stride between channels
#pragma unroll 4
    for (int c = 0; c < Cc; ++c) {
        if (c + 8 < Cc) __builtin_prefetch(wp + (size_t)(c + 8) * cstride, 0, 0);
        float w = wp[(size_t)c * cstride];
#pragma unroll
        for (int b = 0; b < Bn; ++b) acc[b] += fv[b * Cc + c] * w;
    }
#pragma unroll
    for (int b = 0; b < Bn; ++b)
        atomicAdd(&xseg[((size_t)b * NSEG + segs[b]) * DIM + d], acc[b]);
}

// =============================================================================
// 2) global max over segments + concat -> tokens [NTOK, D2] (fp32 residual)
// =============================================================================
__global__ void concat_max_kernel(const float* __restrict__ xseg,
                                  float* __restrict__ xtok) {
    const int b = blockIdx.x;      // 8 blocks
    const int d = threadIdx.x;     // 256 threads
    float m = -INFINITY;
    for (int n = 0; n < NSEG; ++n)
        m = fmaxf(m, xseg[((size_t)b * NSEG + n) * DIM + d]);
    for (int n = 0; n < NSEG; ++n) {
        size_t row = (size_t)b * NSEG + n;
        xtok[row * D2 + d]       = m;
        xtok[row * D2 + DIM + d] = xseg[row * DIM + d];
    }
}

// =============================================================================
// 3) weight packer fp32[K,N] -> f16 in WMMA B-fragment tile layout.
//    tile (tk,tn) of 32(K)x16(N); lane holds 16 halves (one v16h, 32B store).
// =============================================================================
__global__ void pack_weights_kernel(const float* __restrict__ src, int K, int N,
                                    _Float16* __restrict__ dst) {
    const int lane = threadIdx.x;          // 32 threads
    const int lo = lane & 15, hi = lane >> 4;
    const int tK = K >> 5;
    const int tk = blockIdx.x % tK;
    const int tn = blockIdx.x / tK;
    v16h r;
#pragma unroll
    for (int i = 0; i < 16; ++i) {
        int k = tk * 32 + ((i >> 3) * 16 + hi * 8 + (i & 7));
        r[i] = (_Float16)src[(size_t)k * N + tn * 16 + lo];
    }
    *(v16h*)(dst + (((size_t)tn * tK + tk) * 32 + lane) * 16) = r;
}

// =============================================================================
// 3b) per-(b,h) V packer: qkv f16 -> B-fragment tiles (K=NSEG rows, N=DH cols).
//     Gather paid once here instead of 8x inside attn_av.
// =============================================================================
__global__ void pack_v_kernel(const _Float16* __restrict__ qkv,
                              _Float16* __restrict__ vpk) {
    const int lane = threadIdx.x;          // 32 threads
    const int lo = lane & 15, hi = lane >> 4;
    const int tile = blockIdx.x & 31;      // 32 tiles per bh (tK=8, tN=4)
    const int bh   = blockIdx.x >> 5;
    const int b = bh >> 3, h = bh & 7;
    const int tk = tile & 7, tn = tile >> 3;
    const _Float16* Vb = qkv + (size_t)b * NSEG * QS + 2 * INNER + h * DH;
    v16h r;
#pragma unroll
    for (int i = 0; i < 16; ++i) {
        int k = tk * 32 + ((i >> 3) * 16 + hi * 8 + (i & 7));   // m index
        r[i] = Vb[(size_t)k * QS + tn * 16 + lo];
    }
    *(v16h*)(vpk + (((size_t)bh * 32 + tn * 8 + tk) * 32 + lane) * 16) = r;
}

// =============================================================================
// 4) LayerNorm (wave per row of D2=512) -> f16 row-major activation
// =============================================================================
__global__ void layernorm_kernel(const float* __restrict__ x,
                                 const float* __restrict__ g,
                                 const float* __restrict__ bb,
                                 _Float16* __restrict__ y) {
    const int lane = threadIdx.x & 31;
    const int wave = threadIdx.x >> 5;
    const int row  = blockIdx.x * 8 + wave;
    const float* xr = x + (size_t)row * D2;
    float v[16], s = 0.f;
#pragma unroll
    for (int i = 0; i < 16; ++i) { v[i] = xr[lane + i * 32]; s += v[i]; }
#pragma unroll
    for (int off = 16; off >= 1; off >>= 1) s += __shfl_xor(s, off, 32);
    const float mu = s * (1.f / D2);
    float q = 0.f;
#pragma unroll
    for (int i = 0; i < 16; ++i) { float t = v[i] - mu; q += t * t; }
#pragma unroll
    for (int off = 16; off >= 1; off >>= 1) q += __shfl_xor(q, off, 32);
    const float inv = rsqrtf(q * (1.f / D2) + 1e-5f);
    _Float16* yr = y + (size_t)row * D2;
#pragma unroll
    for (int i = 0; i < 16; ++i) {
        int c = lane + i * 32;
        yr[c] = (_Float16)((v[i] - mu) * inv * g[c] + bb[c]);
    }
}

// =============================================================================
// 5) GEMM:  out[M,N] = act( A_f16[M,K] x Bpacked_f16[K,N] + bias (+ resid) )
//    block = 8 waves (4M x 2N), block tile 128x64, wave tile 32x32 (2x2 WMMA)
//    Compile-time epilogue selection -> no per-element branches.
// =============================================================================
template <bool BIAS, bool RESID, bool GELU, bool OUTH>
__global__ void gemm_wmma_kernel(const _Float16* __restrict__ A,
                                 const _Float16* __restrict__ Bp,
                                 const float* __restrict__ bias,
                                 const float* resid, float* outf,
                                 _Float16* outh, int M, int K, int N) {
    const int lane = threadIdx.x & 31;
    const int wave = threadIdx.x >> 5;
    const int lo = lane & 15, hi = lane >> 4;
    const int wm = wave & 3, wn = wave >> 2;
    const int M0 = blockIdx.y * 128 + wm * 32;
    const int N0 = blockIdx.x * 64 + wn * 32;
    const int tK = K >> 5;
    const int tn0 = N0 >> 4;

    v8f acc[2][2] = {};
    for (int kt = 0; kt < tK; ++kt) {
        const int k0 = kt * 32;
        v16h a0 = frag_row_h(A + (size_t)(M0 + lo) * K + k0 + hi * 8);
        v16h a1 = frag_row_h(A + (size_t)(M0 + 16 + lo) * K + k0 + hi * 8);
        v16h b0 = *(const v16h*)(Bp + (((size_t)tn0 * tK + kt) * 32 + lane) * 16);
        v16h b1 = *(const v16h*)(Bp + (((size_t)(tn0 + 1) * tK + kt) * 32 + lane) * 16);
        acc[0][0] = wmma_f16(a0, b0, acc[0][0]);
        acc[0][1] = wmma_f16(a0, b1, acc[0][1]);
        acc[1][0] = wmma_f16(a1, b0, acc[1][0]);
        acc[1][1] = wmma_f16(a1, b1, acc[1][1]);
    }
#pragma unroll
    for (int mi = 0; mi < 2; ++mi)
#pragma unroll
        for (int ni = 0; ni < 2; ++ni) {
            const int row0 = M0 + mi * 16 + hi * 8;
            const int col  = N0 + ni * 16 + lo;
            float bc = 0.f;
            if constexpr (BIAS) bc = bias[col];
#pragma unroll
            for (int r = 0; r < 8; ++r) {
                size_t idx = (size_t)(row0 + r) * N + col;
                float vv = acc[mi][ni][r] + bc;
                if constexpr (RESID) vv += resid[idx];
                if constexpr (GELU)  // exact GELU
                    vv = 0.5f * vv * (1.f + erff(vv * 0.70710678118f));
                if constexpr (OUTH) outh[idx] = (_Float16)vv;
                else                outf[idx] = vv;
            }
        }
}

// =============================================================================
// 6) attention scores:  S[bh,n,m] = scale * Q[n,:] . K[m,:]   (Q,K f16)
//    B-fragment of K^T == row-major fragment of K  -> reuse frag_row_h
// =============================================================================
__global__ void attn_scores_kernel(const _Float16* __restrict__ qkv,
                                   float* __restrict__ scores) {
    const int lane = threadIdx.x & 31;
    const int wave = threadIdx.x >> 5;
    const int lo = lane & 15, hi = lane >> 4;
    const int wm = wave & 3, wn = wave >> 2;
    const int bh = blockIdx.z;
    const int b = bh >> 3, h = bh & 7;
    const int M0 = blockIdx.y * 128 + wm * 32;   // n
    const int N0 = blockIdx.x * 64 + wn * 32;    // m
    const _Float16* Qb = qkv + (size_t)b * NSEG * QS + h * DH;
    const _Float16* Kb = Qb + INNER;

    v8f acc[2][2] = {};
#pragma unroll
    for (int kt = 0; kt < DH / 32; ++kt) {
        const int k0 = kt * 32;
        v16h a0 = frag_row_h(Qb + (size_t)(M0 + lo) * QS + k0 + hi * 8);
        v16h a1 = frag_row_h(Qb + (size_t)(M0 + 16 + lo) * QS + k0 + hi * 8);
        v16h b0 = frag_row_h(Kb + (size_t)(N0 + lo) * QS + k0 + hi * 8);
        v16h b1 = frag_row_h(Kb + (size_t)(N0 + 16 + lo) * QS + k0 + hi * 8);
        acc[0][0] = wmma_f16(a0, b0, acc[0][0]);
        acc[0][1] = wmma_f16(a0, b1, acc[0][1]);
        acc[1][0] = wmma_f16(a1, b0, acc[1][0]);
        acc[1][1] = wmma_f16(a1, b1, acc[1][1]);
    }
    const float scale = 0.125f;  // DH^-0.5
    float* Sb = scores + (size_t)bh * NSEG * NSEG;
#pragma unroll
    for (int mi = 0; mi < 2; ++mi)
#pragma unroll
        for (int ni = 0; ni < 2; ++ni) {
            const int row0 = M0 + mi * 16 + hi * 8;
            const int col  = N0 + ni * 16 + lo;
#pragma unroll
            for (int r = 0; r < 8; ++r)
                Sb[(size_t)(row0 + r) * NSEG + col] = acc[mi][ni][r] * scale;
        }
}

// =============================================================================
// 7) softmax over last dim (256).  wave per row; fp32 in, f16 probs out.
// =============================================================================
__global__ void softmax_kernel(const float* __restrict__ scores,
                               _Float16* __restrict__ probs) {
    const int lane = threadIdx.x & 31;
    const int wave = threadIdx.x >> 5;
    const size_t row = (size_t)blockIdx.x * 8 + wave;   // 16384 rows
    const float* p = scores + row * NSEG;
    _Float16* q = probs + row * NSEG;
    float v[8], mx = -INFINITY;
#pragma unroll
    for (int i = 0; i < 8; ++i) { v[i] = p[lane + i * 32]; mx = fmaxf(mx, v[i]); }
#pragma unroll
    for (int off = 16; off >= 1; off >>= 1) mx = fmaxf(mx, __shfl_xor(mx, off, 32));
    float s = 0.f;
#pragma unroll
    for (int i = 0; i < 8; ++i) { v[i] = __expf(v[i] - mx); s += v[i]; }
#pragma unroll
    for (int off = 16; off >= 1; off >>= 1) s += __shfl_xor(s, off, 32);
    const float inv = 1.f / s;
#pragma unroll
    for (int i = 0; i < 8; ++i) q[lane + i * 32] = (_Float16)(v[i] * inv);
}

// =============================================================================
// 8) attention output:  O[n, h*64+d] = sum_m P[n,m] V[m,d]  (all f16 frags)
// =============================================================================
__global__ void attn_av_kernel(const _Float16* __restrict__ probs,
                               const _Float16* __restrict__ vpk,
                               _Float16* __restrict__ attno) {
    const int lane = threadIdx.x & 31;
    const int wave = threadIdx.x >> 5;
    const int lo = lane & 15, hi = lane >> 4;
    const int wm = wave & 3, wn = wave >> 2;
    const int bh = blockIdx.z;
    const int b = bh >> 3, h = bh & 7;
    const int M0 = blockIdx.y * 128 + wm * 32;   // n
    const int N0 = wn * 32;                      // d (0..63)
    const _Float16* Pb = probs + (size_t)bh * NSEG * NSEG;
    const _Float16* Vp = vpk + (size_t)bh * 32 * 32 * 16;  // 32 tiles * 512 halves
    const int tn0 = N0 >> 4;

    v8f acc[2][2] = {};
    for (int kt = 0; kt < NSEG / 32; ++kt) {
        const int k0 = kt * 32;
        v16h a0 = frag_row_h(Pb + (size_t)(M0 + lo) * NSEG + k0 + hi * 8);
        v16h a1 = frag_row_h(Pb + (size_t)(M0 + 16 + lo) * NSEG + k0 + hi * 8);
        v16h b0 = *(const v16h*)(Vp + (((size_t)tn0 * 8 + kt) * 32 + lane) * 16);
        v16h b1 = *(const v16h*)(Vp + (((size_t)(tn0 + 1) * 8 + kt) * 32 + lane) * 16);
        acc[0][0] = wmma_f16(a0, b0, acc[0][0]);
        acc[0][1] = wmma_f16(a0, b1, acc[0][1]);
        acc[1][0] = wmma_f16(a1, b0, acc[1][0]);
        acc[1][1] = wmma_f16(a1, b1, acc[1][1]);
    }
#pragma unroll
    for (int mi = 0; mi < 2; ++mi)
#pragma unroll
        for (int ni = 0; ni < 2; ++ni) {
            const int row0 = M0 + mi * 16 + hi * 8;
            const int col  = N0 + ni * 16 + lo;         // d within head
#pragma unroll
            for (int r = 0; r < 8; ++r)
                attno[((size_t)b * NSEG + row0 + r) * INNER + h * DH + col] =
                    (_Float16)acc[mi][ni][r];
        }
}

// =============================================================================
// 9) classification head: wave per token, K=512, N=4  (reads f16 LN output)
// =============================================================================
__global__ void head_kernel(const _Float16* __restrict__ hbuf,
                            const float* __restrict__ Wh,
                            const float* __restrict__ bh,
                            float* __restrict__ out) {
    const int lane = threadIdx.x & 31;
    const int wave = threadIdx.x >> 5;
    const int tok = blockIdx.x * 8 + wave;
    const _Float16* xr = hbuf + (size_t)tok * D2;
    float acc[NCLS] = {0.f, 0.f, 0.f, 0.f};
    for (int k = lane; k < D2; k += 32) {
        float xv = (float)xr[k];
#pragma unroll
        for (int c = 0; c < NCLS; ++c) acc[c] += xv * Wh[(size_t)k * NCLS + c];
    }
#pragma unroll
    for (int c = 0; c < NCLS; ++c)
#pragma unroll
        for (int off = 16; off >= 1; off >>= 1)
            acc[c] += __shfl_xor(acc[c], off, 32);
    if (lane == 0)
#pragma unroll
        for (int c = 0; c < NCLS; ++c) out[(size_t)tok * NCLS + c] = acc[c] + bh[c];
}

// =============================================================================
// host orchestration
// =============================================================================
extern "C" void kernel_launch(void* const* d_in, const int* in_sizes, int n_in,
                              void* d_out, int out_size, void* d_ws, size_t ws_size,
                              hipStream_t stream) {
    const float* feature = (const float*)d_in[1];
    const int*   sp      = (const int*)d_in[2];
    const float* W_patch = (const float*)d_in[3];
    const float* b_patch = (const float*)d_in[4];
    const float* ln1_g = (const float*)d_in[5];
    const float* ln1_b = (const float*)d_in[6];
    const float* Wqkv  = (const float*)d_in[7];
    const float* Wo    = (const float*)d_in[8];
    const float* bo    = (const float*)d_in[9];
    const float* ln2_g = (const float*)d_in[10];
    const float* ln2_b = (const float*)d_in[11];
    const float* W1    = (const float*)d_in[12];
    const float* b1    = (const float*)d_in[13];
    const float* W2    = (const float*)d_in[14];
    const float* b2    = (const float*)d_in[15];
    const float* lnf_g = (const float*)d_in[16];
    const float* lnf_b = (const float*)d_in[17];
    const float* W_head = (const float*)d_in[18];
    const float* b_head = (const float*)d_in[19];

    const float* feat2 = feature + 2 * (size_t)Bn * Cc * Ss * Ss;

    // workspace carve-up
    float* ws     = (float*)d_ws;
    float* xseg   = ws;                          // 524288 f32
    float* xtok   = xseg + 524288;               // 1048576 f32 (residual stream)
    float* scores = xtok + 1048576;              // 4194304 f32
    _Float16* hbuf16  = (_Float16*)(scores + 4194304);   // 1048576 f16
    _Float16* qkv16   = hbuf16 + 1048576;        // 3145728 f16
    _Float16* attno16 = qkv16 + 3145728;         // 1048576 f16
    _Float16* mlph16  = attno16 + 1048576;       // 2097152 f16
    _Float16* probs16 = mlph16 + 2097152;        // 4194304 f16
    _Float16* vpk     = probs16 + 4194304;       // 1048576 f16 packed V
    _Float16* wpk     = vpk + 1048576;           // 8388608 f16 packed weights

    const size_t L_QKV = (size_t)D2 * 3 * INNER;   // 786432
    const size_t L_WO  = (size_t)INNER * D2;       // 262144
    const size_t L_W1  = (size_t)D2 * MLPD;        // 524288
    const size_t L_W2  = (size_t)MLPD * D2;        // 524288
    const size_t LSTR  = L_QKV + L_WO + L_W1 + L_W2;

    // pack all transformer weights into WMMA B-fragment layout
    for (int d = 0; d < DEPTH; ++d) {
        _Float16* dst = wpk + d * LSTR;
        pack_weights_kernel<<<(D2 / 32) * (3 * INNER / 16), 32, 0, stream>>>(
            Wqkv + d * L_QKV, D2, 3 * INNER, dst);
        pack_weights_kernel<<<(INNER / 32) * (D2 / 16), 32, 0, stream>>>(
            Wo + d * L_WO, INNER, D2, dst + L_QKV);
        pack_weights_kernel<<<(D2 / 32) * (MLPD / 16), 32, 0, stream>>>(
            W1 + d * L_W1, D2, MLPD, dst + L_QKV + L_WO);
        pack_weights_kernel<<<(MLPD / 32) * (D2 / 16), 32, 0, stream>>>(
            W2 + d * L_W2, MLPD, D2, dst + L_QKV + L_WO + L_W1);
    }

    // patch embedding + segment sum + global-max concat
    init_xseg_kernel<<<2048, 256, 0, stream>>>(xseg, b_patch);
    patch_embed_kernel<<<Ss * Ss, 256, 0, stream>>>(feat2, sp, W_patch, xseg);
    concat_max_kernel<<<Bn, 256, 0, stream>>>(xseg, xtok);

    for (int d = 0; d < DEPTH; ++d) {
        const _Float16* wp = wpk + d * LSTR;
        // --- attention block ---
        layernorm_kernel<<<NTOK / 8, 256, 0, stream>>>(xtok, ln1_g + d * D2,
                                                       ln1_b + d * D2, hbuf16);
        gemm_wmma_kernel<false, false, false, true>
            <<<dim3(3 * INNER / 64, NTOK / 128), 256, 0, stream>>>(
                hbuf16, wp, nullptr, nullptr, nullptr, qkv16, NTOK, D2, 3 * INNER);
        attn_scores_kernel<<<dim3(NSEG / 64, NSEG / 128, Bn * HEADS), 256, 0,
                             stream>>>(qkv16, scores);
        softmax_kernel<<<Bn * HEADS * NSEG / 8, 256, 0, stream>>>(scores, probs16);
        pack_v_kernel<<<Bn * HEADS * 32, 32, 0, stream>>>(qkv16, vpk);
        attn_av_kernel<<<dim3(1, NSEG / 128, Bn * HEADS), 256, 0, stream>>>(
            probs16, vpk, attno16);
        gemm_wmma_kernel<true, true, false, false>
            <<<dim3(D2 / 64, NTOK / 128), 256, 0, stream>>>(
                attno16, wp + L_QKV, bo + d * D2, xtok, xtok, nullptr,
                NTOK, INNER, D2);
        // --- MLP block ---
        layernorm_kernel<<<NTOK / 8, 256, 0, stream>>>(xtok, ln2_g + d * D2,
                                                       ln2_b + d * D2, hbuf16);
        gemm_wmma_kernel<true, false, true, true>
            <<<dim3(MLPD / 64, NTOK / 128), 256, 0, stream>>>(
                hbuf16, wp + L_QKV + L_WO, b1 + d * MLPD, nullptr, nullptr,
                mlph16, NTOK, D2, MLPD);
        gemm_wmma_kernel<true, true, false, false>
            <<<dim3(D2 / 64, NTOK / 128), 256, 0, stream>>>(
                mlph16, wp + L_QKV + L_WO + L_W1, b2 + d * D2, xtok, xtok,
                nullptr, NTOK, MLPD, D2);
    }

    layernorm_kernel<<<NTOK / 8, 256, 0, stream>>>(xtok, lnf_g, lnf_b, hbuf16);
    head_kernel<<<NTOK / 8, 256, 0, stream>>>(hbuf16, W_head, b_head,
                                              (float*)d_out);
}
